// DecoderAutoregCrossAttV2_90658169684691
// MI455X (gfx1250) — matrix operand order repaired and checked by reading, not compiled, not run on hardware
//
#include <hip/hip_runtime.h>
#include <hip/hip_bf16.h>
#include <math.h>

// Problem constants (match reference)
#define B_   8
#define F_   48
#define S_   64
#define D_   512
#define NH_  8
#define L_   4
#define DFF_ 2048
#define MM_  64
#define DH_  64   // D_/NH_

typedef __attribute__((ext_vector_type(16))) __bf16 v16bf;
typedef __attribute__((ext_vector_type(8)))  float  v8f;

// fp32 -> bf16 round-to-nearest-even
__device__ __forceinline__ __bf16 f2bf(float f) {
  unsigned u = __builtin_bit_cast(unsigned, f);
  unsigned r = u + 0x7FFFu + ((u >> 16) & 1u);
  return __builtin_bit_cast(__bf16, (unsigned short)(r >> 16));
}

// ---------------------------------------------------------------------------
// WMMA GEMM:  C[M,N] = act( alpha * A[M,K] @ B + bias + residual (+C) )
//   BW=true : B is weight-style, Bm[n*ldb + k]  (computes A @ W^T)
//   BW=false: B is row-major K x N, Bm[k*ldb + n]
// One wave per 16x32 output tile (A fragment reused across 2 B fragments,
// 2 wmma per barrier pair). LDS tiles are stored in WMMA *fragment order*
// (contiguous k: lane half h holds k = 16h + elem, same for A and B) so each lane
// reads its 16-elem bf16 fragment as contiguous 32B (2x ds_load_b128).
// Bounds handled by clamp+select (no exec-mask branches); K % 32 == 0.
// ---------------------------------------------------------------------------
template <bool BW>
__global__ void wmma_gemm_kernel(const float* __restrict__ A, int lda,
                                 const float* __restrict__ Bm, int ldb,
                                 float* __restrict__ C, int ldc,
                                 const float* __restrict__ bias,
                                 const float* __restrict__ resid, int ldr,
                                 int M, int N, int K, float alpha,
                                 int relu, int accum) {
  __shared__ __align__(32) __bf16 AsF[32][16];      // [lane][elem]
  __shared__ __align__(32) __bf16 BsF[2][32][16];   // [subtile][lane][elem]
  const int lane = threadIdx.x;                     // 0..31 (wave32)
  const int n0 = blockIdx.x * 32;
  const int m0 = blockIdx.y * 16;
  const int row  = lane & 15;
  const int half = lane >> 4;
  v8f acc0 = {}, acc1 = {};

  for (int k0 = 0; k0 < K; k0 += 32) {
    if (k0 + 32 < K) {                              // gfx1250 global_prefetch_b8
      __builtin_prefetch(&A[(size_t)m0 * lda + k0 + 32], 0, 1);
      if (BW) __builtin_prefetch(&Bm[(size_t)n0 * ldb + k0 + 32], 0, 1);
    }
    // ---- A tile 16x32: lane carries k = lane (coalesced rows) ----
    {
      // fragment addr for (m=j, k=lane): flane = j + 16*(k>>4),
      //                                  fe    = k&15
      const int fhalf = lane >> 4;
      const int fe    = lane & 15;
      #pragma unroll
      for (int j = 0; j < 16; ++j) {
        int m  = m0 + j;
        int mc = m < (M - 1) ? m : (M - 1);
        float av = A[(size_t)mc * lda + k0 + lane];
        AsF[j + 16 * fhalf][fe] = f2bf(m < M ? av : 0.f);
      }
    }
    // ---- B tile 32k x 32n in fragment order ----
    if (BW) {
      // lane carries k = lane; loop j over 32 n-columns (coalesced in k)
      const int fhalf = lane >> 4;                  // k>>4
      const int fe    = lane & 15;                  // k&15
      #pragma unroll
      for (int j = 0; j < 32; ++j) {
        int n  = n0 + j;
        int nc = n < (N - 1) ? n : (N - 1);
        float bv = Bm[(size_t)nc * ldb + k0 + lane];
        BsF[j >> 4][(j & 15) + 16 * fhalf][fe] = f2bf(n < N ? bv : 0.f);
      }
    } else {
      // lane carries n = lane (coalesced); loop e over 32 k-rows
      int n  = n0 + lane;
      int nc = n < (N - 1) ? n : (N - 1);
      const int s    = lane >> 4;
      const int ncol = lane & 15;
      #pragma unroll
      for (int e = 0; e < 32; ++e) {
        float bv = Bm[(size_t)(k0 + e) * ldb + nc];
        BsF[s][ncol + 16 * (e >> 4)][e & 15] = f2bf(n < N ? bv : 0.f);
      }
    }
    __syncthreads();
    v16bf af = *reinterpret_cast<const v16bf*>(&AsF[lane][0]);
    v16bf b0 = *reinterpret_cast<const v16bf*>(&BsF[0][lane][0]);
    v16bf b1 = *reinterpret_cast<const v16bf*>(&BsF[1][lane][0]);
    acc0 = __builtin_amdgcn_wmma_f32_16x16x32_bf16(false, af, false, b0,
                                                   (short)0, acc0, false, false);
    acc1 = __builtin_amdgcn_wmma_f32_16x16x32_bf16(false, af, false, b1,
                                                   (short)0, acc1, false, false);
    __syncthreads();
  }
  // Epilogue: C/D layout -> m = r + 8*half, n = subtile*16 + (lane&15)
  #pragma unroll
  for (int s = 0; s < 2; ++s) {
    v8f acc = s ? acc1 : acc0;
    #pragma unroll
    for (int r = 0; r < 8; ++r) {
      int m = m0 + r + 8 * half;
      int n = n0 + s * 16 + row;
      if (m < M && n < N) {
        float v = acc[r] * alpha;
        if (bias)  v += bias[n];
        if (resid) v += resid[(size_t)m * ldr + n];
        if (accum) v += C[(size_t)m * ldc + n];
        if (relu)  v = v > 0.f ? v : 0.f;
        C[(size_t)m * ldc + n] = v;
      }
    }
  }
}

// ---------------------------------------------------------------------------
// Row softmax over 64 columns (sel attention, no mask). One block / row.
// ---------------------------------------------------------------------------
__global__ void softmax64_kernel(float* __restrict__ s) {
  int base = blockIdx.x * 64, tid = threadIdx.x;
  __shared__ float v[64];
  float x = s[base + tid];
  v[tid] = x; __syncthreads();
  float mx = -1e30f;
  for (int j = 0; j < 64; ++j) mx = fmaxf(mx, v[j]);
  __syncthreads();
  float e = __expf(x - mx);
  v[tid] = e; __syncthreads();
  float den = 0.f;
  for (int j = 0; j < 64; ++j) den += v[j];
  s[base + tid] = e / den;
}

// ---------------------------------------------------------------------------
// Incremental causal self-attention with KV cache + ALiBi bias.
// Grid = B*NH blocks, 64 threads (= dh). Query position = `step`.
// ---------------------------------------------------------------------------
__global__ void self_attn_step_kernel(const float* __restrict__ qkv,   // B x 1536
                                      float* __restrict__ kc,          // B*NH*F*DH
                                      float* __restrict__ vc,
                                      float* __restrict__ attn_out,    // B x 512
                                      int step) {
  int b = blockIdx.x / NH_, h = blockIdx.x % NH_;
  int tid = threadIdx.x;                  // 0..63
  __shared__ float sc[F_];
  __shared__ float q_s[DH_];
  float* kch = kc + (size_t)(b * NH_ + h) * F_ * DH_;
  float* vch = vc + (size_t)(b * NH_ + h) * F_ * DH_;
  kch[step * DH_ + tid] = qkv[(size_t)b * 1536 + D_     + h * DH_ + tid];
  vch[step * DH_ + tid] = qkv[(size_t)b * 1536 + 2 * D_ + h * DH_ + tid];
  q_s[tid] = qkv[(size_t)b * 1536 + h * DH_ + tid];
  __syncthreads();
  float slope = 1.0f / (float)(2 << h);   // 2^-(h+1)
  if (tid <= step) {
    float s = 0.f;
    #pragma unroll 4
    for (int d = 0; d < DH_; ++d) s += q_s[d] * kch[tid * DH_ + d];
    sc[tid] = s * 0.125f - slope * (float)((step - tid) / 30);
  }
  __syncthreads();
  float mx = -1e30f;
  for (int j = 0; j <= step; ++j) mx = fmaxf(mx, sc[j]);
  float den = 0.f, o = 0.f;
  for (int j = 0; j <= step; ++j) {
    float p = __expf(sc[j] - mx);
    den += p;
    o += p * vch[j * DH_ + tid];
  }
  attn_out[(size_t)b * D_ + h * DH_ + tid] = o / den;
}

// ---------------------------------------------------------------------------
// LayerNorm over 512 columns; one block (256 thr) per row. y = LN(x)*g + b
// ---------------------------------------------------------------------------
__global__ void layernorm_kernel(const float* __restrict__ x, float* __restrict__ y,
                                 const float* __restrict__ g, const float* __restrict__ bb) {
  int r = blockIdx.x, tid = threadIdx.x;
  __shared__ float red[256];
  const float* xr = x + (size_t)r * D_;
  float s = 0.f;
  for (int c = tid; c < D_; c += 256) s += xr[c];
  red[tid] = s; __syncthreads();
  for (int o = 128; o > 0; o >>= 1) { if (tid < o) red[tid] += red[tid + o]; __syncthreads(); }
  float mu = red[0] * (1.0f / D_); __syncthreads();
  float s2 = 0.f;
  for (int c = tid; c < D_; c += 256) { float d = xr[c] - mu; s2 += d * d; }
  red[tid] = s2; __syncthreads();
  for (int o = 128; o > 0; o >>= 1) { if (tid < o) red[tid] += red[tid + o]; __syncthreads(); }
  float inv = rsqrtf(red[0] * (1.0f / D_) + 1e-5f);
  for (int c = tid; c < D_; c += 256)
    y[(size_t)r * D_ + c] = (xr[c] - mu) * inv * g[c] + bb[c];
}

// ---------------------------------------------------------------------------
// x[b,:] = dec_emb[b, step, :] + PE[step, :]   (sinusoidal, period 30)
// ---------------------------------------------------------------------------
__global__ void embed_pe_kernel(const float* __restrict__ emb, float* __restrict__ x, int step) {
  int b = blockIdx.x, d = threadIdx.x;    // 512 threads
  float pos = (float)(step % 30);
  float ang = pos * __expf(-(float)((d >> 1) * 2) * 0.01798894558f); // ln(1e4)/512
  float pe = (d & 1) ? __cosf(ang) : __sinf(ang);
  x[(size_t)b * D_ + d] = emb[((size_t)b * F_ + step) * D_ + d] + pe;
}

// ---------------------------------------------------------------------------
// Host orchestration
// ---------------------------------------------------------------------------
static inline void gemm(hipStream_t st, const float* A, int lda,
                        const float* Bm, int ldb, float* C, int ldc,
                        const float* bias, const float* res, int ldr,
                        int M, int N, int K, float alpha, int relu, int accum,
                        bool bweight) {
  dim3 g((N + 31) / 32, (M + 15) / 16);
  if (bweight)
    wmma_gemm_kernel<true ><<<g, 32, 0, st>>>(A, lda, Bm, ldb, C, ldc, bias, res, ldr, M, N, K, alpha, relu, accum);
  else
    wmma_gemm_kernel<false><<<g, 32, 0, st>>>(A, lda, Bm, ldb, C, ldc, bias, res, ldr, M, N, K, alpha, relu, accum);
}

extern "C" void kernel_launch(void* const* d_in, const int* in_sizes, int n_in,
                              void* d_out, int out_size, void* d_ws, size_t ws_size,
                              hipStream_t stream) {
  const float* content  = (const float*)d_in[0];   // (B,F,D)
  const float* style    = (const float*)d_in[1];   // (B,D)
  const float* shid     = (const float*)d_in[2];   // (B,S,D)
  const float* istate   = (const float*)d_in[3];   // (B,M)
  const float* ca_in_w  = (const float*)d_in[4];   // (3D,D)
  const float* ca_in_b  = (const float*)d_in[5];
  const float* ca_out_w = (const float*)d_in[6];
  const float* ca_out_b = (const float*)d_in[7];
  const float* se_w     = (const float*)d_in[8];   // (D,2D)
  const float* se_b     = (const float*)d_in[9];
  const float* mm_w     = (const float*)d_in[10];  // (D,M)
  const float* mm_b     = (const float*)d_in[11];
  const float* mmr_w    = (const float*)d_in[12];  // (M,D)
  const float* mmr_b    = (const float*)d_in[13];
  const float* sa_in_w  = (const float*)d_in[14];  // (L,3D,D)
  const float* sa_in_b  = (const float*)d_in[15];
  const float* sa_out_w = (const float*)d_in[16];
  const float* sa_out_b = (const float*)d_in[17];
  const float* xa_in_w  = (const float*)d_in[18];
  const float* xa_in_b  = (const float*)d_in[19];
  const float* xa_out_w = (const float*)d_in[20];
  const float* xa_out_b = (const float*)d_in[21];
  const float* ff1_w    = (const float*)d_in[22];  // (L,DFF,D)
  const float* ff1_b    = (const float*)d_in[23];
  const float* ff2_w    = (const float*)d_in[24];  // (L,D,DFF)
  const float* ff2_b    = (const float*)d_in[25];
  const float* ln_g     = (const float*)d_in[26];  // (L,3,D)
  const float* ln_b     = (const float*)d_in[27];

  float* dec_out = (float*)d_out;                  // (B,F,M)
  float* sel     = (float*)d_out + (size_t)B_ * F_ * MM_;  // (B,F,D)

  float* ws = (float*)d_ws;
  size_t off = 0;
  auto alloc = [&](size_t n) { float* p = ws + off; off += n; return p; };
  float* qh   = alloc((size_t)B_ * F_ * D_);
  float* kh   = alloc((size_t)B_ * S_ * D_);
  float* vh   = alloc((size_t)B_ * S_ * D_);
  float* scor = alloc((size_t)B_ * F_ * S_);
  float* ctx  = alloc((size_t)B_ * F_ * D_);
  float* emb  = alloc((size_t)B_ * F_ * D_);       // dec_emb (B,F,D)
  float* kc   = alloc((size_t)L_ * B_ * NH_ * F_ * DH_);
  float* vc   = alloc((size_t)L_ * B_ * NH_ * F_ * DH_);
  float* xb   = alloc((size_t)B_ * D_);
  float* yb   = alloc((size_t)B_ * D_);
  float* qkvb = alloc((size_t)B_ * 3 * D_);
  float* attb = alloc((size_t)B_ * D_);
  float* vxb  = alloc((size_t)B_ * D_);
  float* hb   = alloc((size_t)B_ * DFF_);
  float* nb   = alloc((size_t)B_ * D_);
  (void)in_sizes; (void)n_in; (void)out_size; (void)ws_size;

  const float inv_sqrt_d = 0.044194173824159216f;  // 1/sqrt(512)
  const size_t lcache = (size_t)B_ * NH_ * F_ * DH_;

  // ---- Phase A: sel = single-head cross-attention(content, style_hiddens) ----
  gemm(stream, content, D_, ca_in_w,                        D_, qh, D_, ca_in_b,          nullptr, 0, B_ * F_, D_, D_, 1.f, 0, 0, true);
  gemm(stream, shid,    D_, ca_in_w + (size_t)D_ * D_,      D_, kh, D_, ca_in_b + D_,     nullptr, 0, B_ * S_, D_, D_, 1.f, 0, 0, true);
  gemm(stream, shid,    D_, ca_in_w + (size_t)2 * D_ * D_,  D_, vh, D_, ca_in_b + 2 * D_, nullptr, 0, B_ * S_, D_, D_, 1.f, 0, 0, true);
  for (int b = 0; b < B_; ++b)   // scores[b] = Q[b] @ K[b]^T / sqrt(D)
    gemm(stream, qh + (size_t)b * F_ * D_, D_, kh + (size_t)b * S_ * D_, D_,
         scor + (size_t)b * F_ * S_, S_, nullptr, nullptr, 0, F_, S_, D_, inv_sqrt_d, 0, 0, true);
  softmax64_kernel<<<B_ * F_, 64, 0, stream>>>(scor);
  for (int b = 0; b < B_; ++b)   // ctx[b] = P[b] @ V[b]  (B row-major K x N)
    gemm(stream, scor + (size_t)b * F_ * S_, S_, vh + (size_t)b * S_ * D_, D_,
         ctx + (size_t)b * F_ * D_, D_, nullptr, nullptr, 0, F_, D_, S_, 1.f, 0, 0, false);
  gemm(stream, ctx, D_, ca_out_w, D_, sel, D_, ca_out_b, nullptr, 0, B_ * F_, D_, D_, 1.f, 0, 0, true);

  // ---- Phase B: initial embedding (position 0, uses sel[:,0]) ----
  gemm(stream, istate, MM_, mm_w, MM_, nb, D_, mm_b, nullptr, 0, B_, D_, MM_, 1.f, 0, 0, true);
  gemm(stream, nb, D_, se_w,      2 * D_, emb, F_ * D_, se_b, style, D_, B_, D_, D_, 1.f, 0, 0, true);
  gemm(stream, sel, F_ * D_, se_w + D_, 2 * D_, emb, F_ * D_, nullptr, nullptr, 0, B_, D_, D_, 1.f, 0, 1, true);

  // ---- Phase C: 48 incremental decode steps (KV-cached; tokens step-invariant) ----
  for (int i = 0; i < F_; ++i) {
    embed_pe_kernel<<<B_, D_, 0, stream>>>(emb, xb, i);
    for (int l = 0; l < L_; ++l) {
      // self-attention
      gemm(stream, xb, D_, sa_in_w + (size_t)l * 3 * D_ * D_, D_, qkvb, 3 * D_,
           sa_in_b + (size_t)l * 3 * D_, nullptr, 0, B_, 3 * D_, D_, 1.f, 0, 0, true);
      self_attn_step_kernel<<<B_ * NH_, DH_, 0, stream>>>(qkvb, kc + (size_t)l * lcache,
                                                          vc + (size_t)l * lcache, attb, i);
      gemm(stream, attb, D_, sa_out_w + (size_t)l * D_ * D_, D_, yb, D_,
           sa_out_b + (size_t)l * D_, xb, D_, B_, D_, D_, 1.f, 0, 0, true);
      layernorm_kernel<<<B_, 256, 0, stream>>>(yb, xb, ln_g + (size_t)(l * 3 + 0) * D_, ln_b + (size_t)(l * 3 + 0) * D_);
      // cross-attention: diagonal mask -> exactly out_w @ (wv @ mem[i] + bv) + biases
      gemm(stream, content + (size_t)i * D_, F_ * D_,
           xa_in_w + (size_t)l * 3 * D_ * D_ + (size_t)2 * D_ * D_, D_, vxb, D_,
           xa_in_b + (size_t)l * 3 * D_ + 2 * D_, nullptr, 0, B_, D_, D_, 1.f, 0, 0, true);
      gemm(stream, vxb, D_, xa_out_w + (size_t)l * D_ * D_, D_, yb, D_,
           xa_out_b + (size_t)l * D_, xb, D_, B_, D_, D_, 1.f, 0, 0, true);
      layernorm_kernel<<<B_, 256, 0, stream>>>(yb, xb, ln_g + (size_t)(l * 3 + 1) * D_, ln_b + (size_t)(l * 3 + 1) * D_);
      // feed-forward
      gemm(stream, xb, D_, ff1_w + (size_t)l * DFF_ * D_, D_, hb, DFF_,
           ff1_b + (size_t)l * DFF_, nullptr, 0, B_, DFF_, D_, 1.f, 1, 0, true);
      gemm(stream, hb, DFF_, ff2_w + (size_t)l * D_ * DFF_, DFF_, yb, D_,
           ff2_b + (size_t)l * D_, xb, D_, B_, D_, DFF_, 1.f, 0, 0, true);
      layernorm_kernel<<<B_, 256, 0, stream>>>(yb, xb, ln_g + (size_t)(l * 3 + 2) * D_, ln_b + (size_t)(l * 3 + 2) * D_);
    }
    // dec_out[:, i, :] = x @ mmr_w^T + mmr_b
    gemm(stream, xb, D_, mmr_w, D_, dec_out + (size_t)i * MM_, F_ * MM_,
         mmr_b, nullptr, 0, B_, MM_, D_, 1.f, 0, 0, true);
    if (i + 1 < F_) {
      // new = dec_out[:,i] @ mm_w^T + mm_b;
      // emb[:,i+1] = new @ se_w[:,:D]^T + sel[:,i] @ se_w[:,D:]^T + se_b + style
      gemm(stream, dec_out + (size_t)i * MM_, F_ * MM_, mm_w, MM_, nb, D_,
           mm_b, nullptr, 0, B_, D_, MM_, 1.f, 0, 0, true);
      gemm(stream, nb, D_, se_w, 2 * D_, emb + (size_t)(i + 1) * D_, F_ * D_,
           se_b, style, D_, B_, D_, D_, 1.f, 0, 0, true);
      gemm(stream, sel + (size_t)i * D_, F_ * D_, se_w + D_, 2 * D_,
           emb + (size_t)(i + 1) * D_, F_ * D_, nullptr, nullptr, 0, B_, D_, D_, 1.f, 0, 1, true);
    }
  }
}
